// GNNActorCritic_2362232013020
// MI455X (gfx1250) — compile-verified, matrix operand order checked by reading, not run on hardware
//
#include <hip/hip_runtime.h>

// GNN actor-critic forward for MI455X (gfx1250, wave32).
// GEMMs via v_wmma_f32_16x16x32_bf16 with pre-swizzled B fragments
// (contiguous v16bf per lane -> global_load_b128, no LDS gathers);
// edge phase via L2-resident gathers (+ global_prefetch) and
// global_atomic_add_f32 (192MB L2 holds the whole ~100MB working set).

#define N_   65536
#define G_   64
#define NPG_ 1024
#define E_   1048576
#define H_   128
#define NF_  3
#define GF_  7
#define ND_  10
#define KP_  160   // actor GEMM K padded 138 -> 160 (multiple of 32)

typedef __attribute__((ext_vector_type(16))) __bf16 v16bf;
typedef __attribute__((ext_vector_type(8)))  float  v8f;
typedef unsigned int u32;

// ---------------------------------------------------------------- utilities

__global__ void gnn_zero_f32(float* p, long n) {
    long i = (long)blockIdx.x * blockDim.x + threadIdx.x;
    long stride = (long)gridDim.x * blockDim.x;
    for (; i < n; i += stride) p[i] = 0.0f;
}

// Pack fp32 weight [din,128] -> bf16 in WMMA B-fragment order:
// chunk = ((nt * (kpad/32) + ks) * 32 + lane), element i in 0..15:
//   k = ks*32 + (i<8 ? hi*8+i : 16 + hi*8 + (i-8)),  n = nt*16 + (lane&15)
// so each lane's per-kstep B fragment is one contiguous 32B v16bf.
__global__ void gnn_pack_w_sw(const float* __restrict__ W, __bf16* __restrict__ out,
                              int din, int kpad) {
    int idx = blockIdx.x * blockDim.x + threadIdx.x;
    if (idx >= kpad * H_) return;
    int i     = idx & 15;
    int chunk = idx >> 4;
    int lane  = chunk & 31;
    int rest  = chunk >> 5;          // nt * ksteps + ks
    int ksteps = kpad >> 5;
    int nt = rest / ksteps;
    int ks = rest - nt * ksteps;
    int rm = lane & 15, hi = lane >> 4;
    int k = ks * 32 + ((i < 8) ? (hi * 8 + i) : (16 + hi * 8 + (i - 8)));
    int n = nt * 16 + rm;
    float v = (k < din) ? W[k * H_ + n] : 0.0f;
    out[idx] = (__bf16)v;
}

// ------------------------------------------------------------- WMMA GEMM
// out[N,128] = A[N,K](bf16, row stride KP_) @ Wsw(K,128 swizzled) + bias
// block: 256 threads = 8 waves; block tile 16 rows x 128 cols.
// A tile staged in LDS (b128 copies, b128 fragment loads); B fragments
// loaded directly from L2-resident swizzled global weights as v16bf.

__global__ __launch_bounds__(256) void gnn_wmma_gemm(
    const __bf16* __restrict__ A, const __bf16* __restrict__ Wsw,
    const float* __restrict__ bias, float* __restrict__ out, int K) {
    __shared__ __bf16 sA[16 * KP_];

    const int tid  = threadIdx.x;
    const int wave = tid >> 5;
    const int lane = tid & 31;
    const int m0   = blockIdx.x * 16;

    // Stage A tile (16 x K) into LDS with 16B copies
    const int rowChunks = K >> 3;            // uint4 per row
    for (int r = wave; r < 16; r += 8) {
        const uint4* src = (const uint4*)(A + (size_t)(m0 + r) * KP_);
        uint4* dst = (uint4*)(sA + r * K);
        for (int i = lane; i < rowChunks; i += 32) dst[i] = src[i];
    }
    __syncthreads();

    const int rm = lane & 15;
    const int hi = lane >> 4;
    const int ksteps = K >> 5;
    const __bf16* bp = Wsw + ((size_t)(wave * ksteps) * 32 + lane) * 16;
    const __bf16* ap0 = sA + rm * K + hi * 8;

    v8f acc = {0.f, 0.f, 0.f, 0.f, 0.f, 0.f, 0.f, 0.f};
    #pragma unroll 5
    for (int ks = 0; ks < ksteps; ++ks) {
        v16bf a, b;
        const __bf16* ap = ap0 + ks * 32;
        ((uint4*)&a)[0] = *(const uint4*)(ap);        // K = k0 + hi*8 + 0..7
        ((uint4*)&a)[1] = *(const uint4*)(ap + 16);   // K = k0+16+hi*8 + 0..7
        b = *(const v16bf*)(bp + (size_t)ks * 32 * 16);
        acc = __builtin_amdgcn_wmma_f32_16x16x32_bf16(
            false, a, false, b, (short)0, acc, false, false);
    }

    // C layout: VGPR r, lanes 0-15 -> M=r, lanes 16-31 -> M=r+8; N=lane&15.
    const int c = wave * 16 + rm;
    const float bc = bias[c];
    #pragma unroll
    for (int r = 0; r < 8; ++r) {
        int m = m0 + r + hi * 8;
        out[(size_t)m * H_ + c] = acc[r] + bc;
    }
}

// ---------------------------------------------------------------- conv0

__global__ __launch_bounds__(256) void gnn_conv0_edge(
    const float* __restrict__ x, const float* __restrict__ gf,
    const int* __restrict__ batch, const int* __restrict__ ei,
    const float* __restrict__ ea, const float* __restrict__ eW,
    const float* __restrict__ eb, float* __restrict__ agg) {
    int e = blockIdx.x * 256 + threadIdx.x;
    if (e >= E_) return;
    int src = ei[e], dst = ei[E_ + e];
    float a = ea[e];
    int bs = batch[src];
    #pragma unroll
    for (int d = 0; d < ND_; ++d) {
        float hs = (d < NF_) ? x[src * NF_ + d] : gf[bs * GF_ + d - NF_];
        float m = fmaxf(hs + a * eW[d] + eb[d], 0.0f);
        atomicAdd(&agg[(size_t)dst * ND_ + d], m);
    }
}

// h=(1+eps)*h0+agg; t=h@W1+b1; y=relu(LN(t)); act[n,KP_] (bf16)
__global__ __launch_bounds__(128) void gnn_conv0_node(
    const float* __restrict__ x, const float* __restrict__ gf,
    const int* __restrict__ batch, const float* __restrict__ agg0,
    const float* __restrict__ eps, const float* __restrict__ W1,
    const float* __restrict__ b1, const float* __restrict__ lng,
    const float* __restrict__ lnb, __bf16* __restrict__ act) {
    __shared__ float sh[ND_];
    __shared__ float s1[H_], s2[H_];
    int n = blockIdx.x, c = threadIdx.x;
    if (c < ND_) {
        float h0 = (c < NF_) ? x[n * NF_ + c] : gf[batch[n] * GF_ + c - NF_];
        sh[c] = (1.0f + eps[0]) * h0 + agg0[(size_t)n * ND_ + c];
    }
    __syncthreads();
    float t = b1[c];
    #pragma unroll
    for (int k = 0; k < ND_; ++k) t += sh[k] * W1[k * H_ + c];
    s1[c] = t; s2[c] = t * t;
    __syncthreads();
    for (int off = 64; off > 0; off >>= 1) {
        if (c < off) { s1[c] += s1[c + off]; s2[c] += s2[c + off]; }
        __syncthreads();
    }
    float mu  = s1[0] * (1.0f / H_);
    float var = s2[0] * (1.0f / H_) - mu * mu;
    float y = (t - mu) * rsqrtf(var + 1e-5f) * lng[c] + lnb[c];
    act[(size_t)n * KP_ + c] = (__bf16)fmaxf(y, 0.0f);
}

// ---------------------------------------------------------------- conv1

__global__ void gnn_relu_inplace(float* p, long n) {
    long i = (long)blockIdx.x * blockDim.x + threadIdx.x;
    if (i < n) p[i] = fmaxf(p[i], 0.0f);
}

// one wave per 4 edges, 4 channels/lane; prefetch gathered rows
// (global_prefetch_b8) then gather + global_atomic_add_f32 scatter.
__global__ __launch_bounds__(256) void gnn_conv1_edge(
    const float* __restrict__ h1, const int* __restrict__ ei,
    const float* __restrict__ ea, const float* __restrict__ eW,
    const float* __restrict__ eb, float* __restrict__ agg) {
    int wave = threadIdx.x >> 5, lane = threadIdx.x & 31;
    long e0 = ((long)blockIdx.x * 8 + wave) * 4;
    if (e0 >= E_) return;
    const float4 wv = *(const float4*)(eW + lane * 4);
    const float4 bv = *(const float4*)(eb + lane * 4);
    int srcs[4], dsts[4];
    float as[4];
    #pragma unroll
    for (int j = 0; j < 4; ++j) {
        srcs[j] = ei[e0 + j];
        dsts[j] = ei[E_ + e0 + j];
        as[j]   = ea[e0 + j];
    }
    #pragma unroll
    for (int j = 0; j < 4; ++j)
        __builtin_prefetch(h1 + (size_t)srcs[j] * H_ + lane * 4, 0, 3);
    #pragma unroll
    for (int j = 0; j < 4; ++j) {
        const float4 hv = *(const float4*)(h1 + (size_t)srcs[j] * H_ + lane * 4);
        float a = as[j];
        float4 m;
        m.x = fmaxf(hv.x + a * wv.x + bv.x, 0.0f);
        m.y = fmaxf(hv.y + a * wv.y + bv.y, 0.0f);
        m.z = fmaxf(hv.z + a * wv.z + bv.z, 0.0f);
        m.w = fmaxf(hv.w + a * wv.w + bv.w, 0.0f);
        float* p = agg + (size_t)dsts[j] * H_ + lane * 4;
        atomicAdd(p + 0, m.x);
        atomicAdd(p + 1, m.y);
        atomicAdd(p + 2, m.z);
        atomicAdd(p + 3, m.w);
    }
}

__global__ void gnn_node_update(const float* __restrict__ h1,
                                const float* __restrict__ agg,
                                const float* __restrict__ eps,
                                __bf16* __restrict__ act) {
    long i = (long)blockIdx.x * 256 + threadIdx.x;
    if (i >= (long)N_ * H_) return;
    float u = (1.0f + eps[0]) * h1[i] + agg[i];
    act[(size_t)(i >> 7) * KP_ + (i & 127)] = (__bf16)u;
}

__global__ __launch_bounds__(128) void gnn_ln_relu(
    const float* __restrict__ t_in, const float* __restrict__ lng,
    const float* __restrict__ lnb, __bf16* __restrict__ act) {
    __shared__ float s1[H_], s2[H_];
    int n = blockIdx.x, c = threadIdx.x;
    float t = t_in[(size_t)n * H_ + c];
    s1[c] = t; s2[c] = t * t;
    __syncthreads();
    for (int off = 64; off > 0; off >>= 1) {
        if (c < off) { s1[c] += s1[c + off]; s2[c] += s2[c + off]; }
        __syncthreads();
    }
    float mu  = s1[0] * (1.0f / H_);
    float var = s2[0] * (1.0f / H_) - mu * mu;
    float y = (t - mu) * rsqrtf(var + 1e-5f) * lng[c] + lnb[c];
    act[(size_t)n * KP_ + c] = (__bf16)fmaxf(y, 0.0f);
}

// relu in place (keep f32 h2 for critic) and pack bf16 into actor input cols 0..127
__global__ void gnn_relu_to_act(float* __restrict__ f, __bf16* __restrict__ act) {
    long i = (long)blockIdx.x * 256 + threadIdx.x;
    if (i >= (long)N_ * H_) return;
    float v = fmaxf(f[i], 0.0f);
    f[i] = v;
    act[(size_t)(i >> 7) * KP_ + (i & 127)] = (__bf16)v;
}

// ---------------------------------------------------------------- heads

// actor input cols 128..159: [xg(7), x[:, :3](3), zeros(22)]
__global__ void gnn_actor_extra(const float* __restrict__ x,
                                const float* __restrict__ gf,
                                const int* __restrict__ batch,
                                __bf16* __restrict__ act) {
    int i = blockIdx.x * 256 + threadIdx.x;
    if (i >= N_ * 32) return;
    int n = i >> 5, j = i & 31;
    float v = 0.0f;
    if (j < GF_)            v = gf[batch[n] * GF_ + j];
    else if (j < GF_ + NF_) v = x[n * NF_ + (j - GF_)];
    act[(size_t)n * KP_ + H_ + j] = (__bf16)v;
}

__global__ __launch_bounds__(256) void gnn_actor_head(
    const float* __restrict__ a, const float* __restrict__ w2,
    const float* __restrict__ b2, float* __restrict__ logits) {
    int wave = threadIdx.x >> 5, lane = threadIdx.x & 31;
    int n = blockIdx.x * 8 + wave;
    if (n >= N_) return;
    float s = 0.0f;
    #pragma unroll
    for (int j = 0; j < 4; ++j) {
        int c = lane + 32 * j;
        s += fmaxf(a[(size_t)n * H_ + c], 0.0f) * w2[c];
    }
    for (int off = 16; off > 0; off >>= 1) s += __shfl_down(s, off, 32);
    if (lane == 0) logits[n] = s + b2[0];
}

__global__ __launch_bounds__(128) void gnn_critic_pool(
    const float* __restrict__ h2, const float* __restrict__ gf,
    float* __restrict__ pooled) {
    int g = blockIdx.x, c = threadIdx.x;
    const float* base = h2 + (size_t)g * NPG_ * H_ + c;
    float s = 0.0f, mx = -3.4e38f;
    for (int r = 0; r < NPG_; ++r) {
        float v = base[(size_t)r * H_];
        s += v;
        mx = fmaxf(mx, v);
    }
    pooled[g * 263 + c]      = s * (1.0f / NPG_);
    pooled[g * 263 + H_ + c] = mx;
    if (c < GF_) pooled[g * 263 + 2 * H_ + c] = gf[g * GF_ + c];
}

__global__ __launch_bounds__(128) void gnn_critic_mlp(
    const float* __restrict__ pooled, const float* __restrict__ W1,
    const float* __restrict__ b1, const float* __restrict__ w2,
    const float* __restrict__ b2, float* __restrict__ value) {
    __shared__ float red[H_];
    int g = blockIdx.x, c = threadIdx.x;
    float s = b1[c];
    for (int k = 0; k < 263; ++k) s += pooled[g * 263 + k] * W1[k * H_ + c];
    red[c] = fmaxf(s, 0.0f) * w2[c];
    __syncthreads();
    for (int off = 64; off > 0; off >>= 1) {
        if (c < off) red[c] += red[c + off];
        __syncthreads();
    }
    if (c == 0) value[g] = red[0] + b2[0];
}

// ---------------------------------------------------------------- launch

extern "C" void kernel_launch(void* const* d_in, const int* in_sizes, int n_in,
                              void* d_out, int out_size, void* d_ws, size_t ws_size,
                              hipStream_t stream) {
    // setup_inputs order + jax pytree (sorted dict keys) flatten of params:
    const float* x    = (const float*)d_in[0];
    const int*   ei   = (const int*)d_in[1];
    const float* ea   = (const float*)d_in[2];
    const int*   batch= (const int*)d_in[3];
    const float* gf   = (const float*)d_in[4];
    const float* a1W  = (const float*)d_in[5];   // actor1.W [138,128]
    const float* a1b  = (const float*)d_in[6];
    const float* a2W  = (const float*)d_in[7];   // actor2.W [128,1]
    const float* a2b  = (const float*)d_in[8];
    const float* c0eW = (const float*)d_in[9];   // conv0.edge.W [1,10]
    const float* c0eb = (const float*)d_in[10];
    const float* c0eps= (const float*)d_in[11];
    const float* c0W1 = (const float*)d_in[12];  // conv0.lin1.W [10,128]
    const float* c0b1 = (const float*)d_in[13];
    const float* c0W2 = (const float*)d_in[14];  // conv0.lin2.W [128,128]
    const float* c0b2 = (const float*)d_in[15];
    const float* c0lnb= (const float*)d_in[16];
    const float* c0lng= (const float*)d_in[17];
    const float* c1eW = (const float*)d_in[18];  // conv1.edge.W [1,128]
    const float* c1eb = (const float*)d_in[19];
    const float* c1eps= (const float*)d_in[20];
    const float* c1W1 = (const float*)d_in[21];
    const float* c1b1 = (const float*)d_in[22];
    const float* c1W2 = (const float*)d_in[23];
    const float* c1b2 = (const float*)d_in[24];
    const float* c1lnb= (const float*)d_in[25];
    const float* c1lng= (const float*)d_in[26];
    const float* cr1W = (const float*)d_in[27];  // critic1.W [263,128]
    const float* cr1b = (const float*)d_in[28];
    const float* cr2W = (const float*)d_in[29];
    const float* cr2b = (const float*)d_in[30];

    float* logits = (float*)d_out;         // [N]
    float* value  = (float*)d_out + N_;    // [G]

    // workspace layout
    char* ws = (char*)d_ws;
    const size_t NB = (size_t)N_ * H_ * sizeof(float);           // 32 MB
    float*  F1  = (float*)ws;                                    // [N,128]
    float*  F2  = (float*)(ws + NB);                             // [N,128]
    __bf16* ACT = (__bf16*)(ws + 2 * NB);                        // [N,160]
    __bf16* WB  = (__bf16*)(ws + 2 * NB + (size_t)N_ * KP_ * 2); // swizzled weights
    __bf16* W02 = WB;                       // conv0.lin2 [128,128]
    __bf16* W11 = WB + 128 * 128;           // conv1.lin1
    __bf16* W12 = WB + 2 * 128 * 128;       // conv1.lin2
    __bf16* WA1 = WB + 3 * 128 * 128;       // actor1 [160,128] (padded)
    float*  POOL = (float*)(WB + 3 * 128 * 128 + KP_ * 128);     // [64,263]

    const int T = 256;

    // pack weights into bf16 B-fragment swizzled order
    gnn_pack_w_sw<<<(128 * 128 + T - 1) / T, T, 0, stream>>>(c0W2, W02, 128, 128);
    gnn_pack_w_sw<<<(128 * 128 + T - 1) / T, T, 0, stream>>>(c1W1, W11, 128, 128);
    gnn_pack_w_sw<<<(128 * 128 + T - 1) / T, T, 0, stream>>>(c1W2, W12, 128, 128);
    gnn_pack_w_sw<<<(KP_ * 128 + T - 1) / T, T, 0, stream>>>(a1W, WA1, 138, KP_);

    // ---- conv0 ----
    gnn_zero_f32<<<4096, T, 0, stream>>>(F2, (long)N_ * ND_);
    gnn_conv0_edge<<<E_ / T, T, 0, stream>>>(x, gf, batch, ei, ea, c0eW, c0eb, F2);
    gnn_conv0_node<<<N_, 128, 0, stream>>>(x, gf, batch, F2, c0eps, c0W1, c0b1,
                                           c0lng, c0lnb, ACT);
    gnn_wmma_gemm<<<N_ / 16, T, 0, stream>>>(ACT, W02, c0b2, F1, 128);
    gnn_relu_inplace<<<(int)(((long)N_ * H_ + T - 1) / T), T, 0, stream>>>(F1, (long)N_ * H_);

    // ---- conv1 ----
    gnn_zero_f32<<<8192, T, 0, stream>>>(F2, (long)N_ * H_);
    gnn_conv1_edge<<<E_ / 32, T, 0, stream>>>(F1, ei, ea, c1eW, c1eb, F2);
    gnn_node_update<<<(int)(((long)N_ * H_ + T - 1) / T), T, 0, stream>>>(F1, F2, c1eps, ACT);
    gnn_wmma_gemm<<<N_ / 16, T, 0, stream>>>(ACT, W11, c1b1, F1, 128);
    gnn_ln_relu<<<N_, 128, 0, stream>>>(F1, c1lng, c1lnb, ACT);
    gnn_wmma_gemm<<<N_ / 16, T, 0, stream>>>(ACT, W12, c1b2, F2, 128);
    gnn_relu_to_act<<<(int)(((long)N_ * H_ + T - 1) / T), T, 0, stream>>>(F2, ACT);

    // ---- actor ----
    gnn_actor_extra<<<(N_ * 32 + T - 1) / T, T, 0, stream>>>(x, gf, batch, ACT);
    gnn_wmma_gemm<<<N_ / 16, T, 0, stream>>>(ACT, WA1, a1b, F1, KP_);
    gnn_actor_head<<<N_ / 8, T, 0, stream>>>(F1, a2W, a2b, logits);

    // ---- critic ----
    gnn_critic_pool<<<G_, 128, 0, stream>>>(F2, gf, POOL);
    gnn_critic_mlp<<<G_, 128, 0, stream>>>(POOL, cr1W, cr1b, cr2W, cr2b, value);
}
